// DSMoE_43508018709051
// MI455X (gfx1250) — matrix operand order repaired and checked by reading, not compiled, not run on hardware
//
#include <hip/hip_runtime.h>
#include <math.h>

// ---------------------------------------------------------------------------
// DeepSeek-style MoE (top-2 of 8 routed experts + 1 shared expert), gfx1250.
//   * top-2 + shared only: ~77 GFLOP vs 232 GFLOP dense-combine reference.
//   * GEMMs on v_wmma_f32_16x16x32_f16 (f16 in, f32 accumulate).
//   * Weights transposed once during f32->f16 convert so WMMA A and B
//     fragments are contiguous LDS reads.
//   * Weight / activation tiles staged by the Tensor Data Mover
//     (tensor_load_to_lds, TENSORcnt), double-buffered.
//   * Deterministic: no float atomics; per-(token,k) slot ids recorded, final
//     combine gathers per-slot GEMM outputs.
// ---------------------------------------------------------------------------

typedef __attribute__((ext_vector_type(16))) _Float16 v16h;
typedef __attribute__((ext_vector_type(8)))  float    v8f;
typedef __attribute__((ext_vector_type(4)))  unsigned v4u;
typedef __attribute__((ext_vector_type(8)))  int      v8i;
typedef __attribute__((ext_vector_type(4)))  int      v4i;

#define T_TOK 8192
#define DIMSZ 512
#define HIDSZ 1024
#define NEXP  8
#define TOPK  2

// ------------------------- TDM 2D tile load helper -------------------------
// Builds a D# (groups 0/1 per CDNA5 ISA 8.3/8.4) for a 2D tile of 16-bit
// elements and issues TENSOR_LOAD_TO_LDS. Dims are "remaining extents" from
// the tile start so rows/cols past the logical tensor read back zeros.
__device__ __forceinline__ void tdm_load_2d(unsigned lds_off, const void* gptr,
                                            unsigned tdim0, unsigned tdim1,
                                            unsigned stride0,
                                            unsigned tile0, unsigned tile1) {
  unsigned long long ga = (unsigned long long)(size_t)gptr;
  v4u g0;
  g0.x = 1u;                                   // count=1, user descriptor
  g0.y = lds_off;                              // LDS byte address
  g0.z = (unsigned)(ga & 0xFFFFFFFFu);         // global_addr[31:0]
  g0.w = (unsigned)((ga >> 32) & 0x01FFFFFFu)  // global_addr[56:32]
         | (2u << 30);                         // type=2 (image)
  v8i g1;
  g1[0] = (int)(1u << 16);                                       // data_size=2B
  g1[1] = (int)((tdim0 & 0xFFFFu) << 16);                        // tensor_dim0 lo
  g1[2] = (int)(((tdim0 >> 16) & 0xFFFFu) | ((tdim1 & 0xFFFFu) << 16));
  g1[3] = (int)(((tdim1 >> 16) & 0xFFFFu) | ((tile0 & 0xFFFFu) << 16));
  g1[4] = (int)(tile1 & 0xFFFFu);                                // tile_dim1 (tile_dim2=0)
  g1[5] = (int)stride0;                                          // dim0 stride lo32
  g1[6] = 0;                                                     // stride hi, stride1 lo
  g1[7] = 0;
  v4i z4; z4[0] = 0; z4[1] = 0; z4[2] = 0; z4[3] = 0;
  v8i z8; z8[0] = 0; z8[1] = 0; z8[2] = 0; z8[3] = 0;
  z8[4] = 0; z8[5] = 0; z8[6] = 0; z8[7] = 0;
  // 6-arg toolchain signature: (g0, g1, g2, g3, g4, cpol)
  __builtin_amdgcn_tensor_load_to_lds(g0, g1, z4, z4, z8, 0);
}

__device__ __forceinline__ unsigned lds_addr_of(const void* p) {
  return (unsigned)(size_t)p;  // low 32 bits of a generic LDS pointer = LDS offset
}

// -------------------------- fp32 -> f16 convert ----------------------------
__global__ void cvt_f32_f16(const float* __restrict__ src,
                            _Float16* __restrict__ dst, int n) {
  int i = blockIdx.x * blockDim.x + threadIdx.x;
  if (i < n) dst[i] = (_Float16)src[i];
}

// batched tiled transpose + convert: dst[b][c][r] = (f16)src[b][r][c]
__global__ void transpose_cvt(const float* __restrict__ src,
                              _Float16* __restrict__ dst, int R, int C) {
  __shared__ _Float16 tile[32][33];
  const int b = blockIdx.z;
  const float* S = src + (size_t)b * R * C;
  _Float16* Dp = dst + (size_t)b * R * C;
  const int c0 = blockIdx.x * 32, r0 = blockIdx.y * 32;
  const int x = threadIdx.x, y = threadIdx.y;  // 32 x 8
#pragma unroll
  for (int j = 0; j < 32; j += 8)
    tile[y + j][x] = (_Float16)S[(size_t)(r0 + y + j) * C + c0 + x];
  __syncthreads();
#pragma unroll
  for (int j = 0; j < 32; j += 8)
    Dp[(size_t)(c0 + y + j) * R + r0 + x] = tile[x][y + j];
}

__global__ void init_counts(int* counts) {
  if (threadIdx.x < NEXP) counts[threadIdx.x] = 0;
}

// ------------------------------- router ------------------------------------
__global__ void router_kernel(const float* __restrict__ x,
                              const float* __restrict__ gw,
                              const float* __restrict__ bias,
                              int* __restrict__ idx,
                              float* __restrict__ gates,
                              int* __restrict__ counts) {
  int t = blockIdx.x * blockDim.x + threadIdx.x;
  if (t >= T_TOK) return;
  float s[NEXP];
#pragma unroll
  for (int e = 0; e < NEXP; ++e) s[e] = 0.f;
  const float* xr = x + (size_t)t * DIMSZ;
  for (int d = 0; d < DIMSZ; ++d) {
    float xv = xr[d];
#pragma unroll
    for (int e = 0; e < NEXP; ++e) s[e] += xv * gw[e * DIMSZ + d];
  }
  int i0 = 0, i1 = 0;
  float b0 = -1e30f, b1 = -1e30f;
#pragma unroll
  for (int e = 0; e < NEXP; ++e) {     // biased top-2 -> indices
    float v = s[e] + bias[e];
    if (v > b0) { b1 = b0; i1 = i0; b0 = v; i0 = e; }
    else if (v > b1) { b1 = v; i1 = e; }
  }
  float v0 = -1e30f, v1 = -1e30f;
#pragma unroll
  for (int e = 0; e < NEXP; ++e) {     // unbiased top-2 values -> gates
    float v = s[e];
    if (v > v0) { v1 = v0; v0 = v; }
    else if (v > v1) { v1 = v; }
  }
  float p0 = 1.f / (1.f + __expf(-v0));
  float p1 = 1.f / (1.f + __expf(-v1));
  float inv = 1.f / (p0 + p1);
  idx[2 * t] = i0;         idx[2 * t + 1] = i1;
  gates[2 * t] = p0 * inv; gates[2 * t + 1] = p1 * inv;
  atomicAdd(&counts[i0], 1);
  atomicAdd(&counts[i1], 1);
}

__global__ void scan_kernel(const int* __restrict__ counts,
                            int* __restrict__ offs, int* __restrict__ cursor) {
  if (threadIdx.x == 0 && blockIdx.x == 0) {
    int off = 0;
    for (int e = 0; e < NEXP; ++e) { offs[e] = off; cursor[e] = off; off += counts[e]; }
    offs[NEXP] = off;
  }
}

__global__ void scatter_kernel(const int* __restrict__ idx,
                               int* __restrict__ cursor,
                               int* __restrict__ bucket_tok,
                               int* __restrict__ slot_of) {
  int t = blockIdx.x * blockDim.x + threadIdx.x;
  if (t >= T_TOK) return;
#pragma unroll
  for (int k = 0; k < TOPK; ++k) {
    int e = idx[2 * t + k];
    int pos = atomicAdd(&cursor[e], 1);
    bucket_tok[pos] = t;
    slot_of[2 * t + k] = pos;
  }
}

// --------------------------- WMMA fragment helpers -------------------------
// A 16x32 (f16): lane L -> row m=L%16, halves j: K=j+16*(L/16)
//   => two contiguous 8-half chunks at half-offsets {ks+16g, ks+16g+8}.
// B 32x16 (f16): lane L -> col n=L%16, halves j: K=j+16*(L/16)
//   => one contiguous 16-half chunk at half-offset ks+16g (weights pre-transposed).
// C/D f32: VGPR r -> M=r+8*(L/16), N=L%16.
union FragA { v16h v; uint4 q[2]; };

#define WMMA16(ACC, A, B) \
  ACC = __builtin_amdgcn_wmma_f32_16x16x32_f16(false, A, false, B, (short)0, ACC, false, false)

// ------------------------------- GEMM1 -------------------------------------
// h = silu(X@W1) * (X@W3); X rows gathered via bucket list (identity for the
// shared expert, blockIdx.z == 8). Tiles: 64(M) x 64(N) x 64(K), B1/B3 tiles
// TDM-loaded (pre-transposed [N][K] layout), A staged manually (gathered).
__global__ __launch_bounds__(256)
void gemm1_kernel(const _Float16* __restrict__ xh,
                  const _Float16* __restrict__ w1t,
                  const _Float16* __restrict__ w3t,
                  const _Float16* __restrict__ sw1t,
                  const _Float16* __restrict__ sw3t,
                  const int* __restrict__ counts,
                  const int* __restrict__ offs,
                  const int* __restrict__ bucket_tok,
                  _Float16* __restrict__ h_rt,
                  _Float16* __restrict__ h_sh) {
  const int e = blockIdx.z;
  const bool is_shared = (e == NEXP);
  const int cnt = is_shared ? T_TOK : counts[e];
  const int m0 = blockIdx.y * 64;
  if (m0 >= cnt) return;
  const int n0 = blockIdx.x * 64;
  const int rowbase = is_shared ? 0 : offs[e];
  const _Float16* W1 = is_shared ? sw1t : (w1t + (size_t)e * HIDSZ * DIMSZ);  // [H][D]
  const _Float16* W3 = is_shared ? sw3t : (w3t + (size_t)e * HIDSZ * DIMSZ);
  _Float16* Hout = is_shared ? h_sh : h_rt;

  __shared__ int tokLds[64];
  __shared__ __align__(16) _Float16 aLds[2][64][64];      // [buf][m][k]
  __shared__ __align__(16) _Float16 bLds[2][2][64][64];   // [buf][mat][n][k]

  const int tid = threadIdx.x;
  if (tid < 64) {
    int m = m0 + tid;
    tokLds[tid] = (m < cnt) ? (is_shared ? m : bucket_tok[rowbase + m]) : 0;
  }
  __syncthreads();

  const int lane = tid & 31;
  const int wv = tid >> 5;
  const int mrow = wv >> 1;       // 16-row band 0..3
  const int nc0 = (wv & 1) * 2;   // first of two 16-col bands
  const int g = lane >> 4;
  const int lm = lane & 15;

  v8f acc10 = {}, acc11 = {}, acc30 = {}, acc31 = {};

  const int NCH = DIMSZ / 64;

  auto issueB = [&](int kc, int buf) {
    int k0 = kc * 64;
    tdm_load_2d(lds_addr_of(&bLds[buf][0][0][0]), W1 + (size_t)n0 * DIMSZ + k0,
                DIMSZ - k0, HIDSZ - n0, DIMSZ, 64, 64);
    tdm_load_2d(lds_addr_of(&bLds[buf][1][0][0]), W3 + (size_t)n0 * DIMSZ + k0,
                DIMSZ - k0, HIDSZ - n0, DIMSZ, 64, 64);
  };
  auto stageA = [&](int kc, int buf) {
    int k0 = kc * 64;
#pragma unroll
    for (int it = 0; it < 2; ++it) {          // 64 rows x 64 halves = 512 uint4
      int q = tid + it * 256;
      int row = q >> 3, c4 = q & 7;
      const uint4* s = (const uint4*)(xh + (size_t)tokLds[row] * DIMSZ + k0) + c4;
      *(uint4*)&aLds[buf][row][c4 * 8] = *s;
    }
  };

  if (wv == 0) issueB(0, 0);
  stageA(0, 0);

  for (int i = 0; i < NCH; ++i) {
    const int cur = i & 1;
    if (i + 1 < NCH) {
      if (wv == 0) issueB(i + 1, cur ^ 1);
      stageA(i + 1, cur ^ 1);
      __builtin_amdgcn_s_wait_tensorcnt((short)2);
    } else {
      __builtin_amdgcn_s_wait_tensorcnt((short)0);
    }
    __syncthreads();  // chunk i: B (TDM) + A (ds stores) visible to all waves

#pragma unroll
    for (int ks = 0; ks < 64; ks += 32) {
      FragA a, b;
      a.q[0] = *(const uint4*)&aLds[cur][mrow * 16 + lm][ks + 16 * g];
      a.q[1] = *(const uint4*)&aLds[cur][mrow * 16 + lm][ks + 16 * g + 8];
      const int nA = nc0 * 16 + lm;
      const int nB = nA + 16;
      b.q[0] = *(const uint4*)&bLds[cur][0][nA][ks + 16 * g];
      b.q[1] = *(const uint4*)&bLds[cur][0][nA][ks + 16 * g + 8];
      WMMA16(acc10, a.v, b.v);
      b.q[0] = *(const uint4*)&bLds[cur][0][nB][ks + 16 * g];
      b.q[1] = *(const uint4*)&bLds[cur][0][nB][ks + 16 * g + 8];
      WMMA16(acc11, a.v, b.v);
      b.q[0] = *(const uint4*)&bLds[cur][1][nA][ks + 16 * g];
      b.q[1] = *(const uint4*)&bLds[cur][1][nA][ks + 16 * g + 8];
      WMMA16(acc30, a.v, b.v);
      b.q[0] = *(const uint4*)&bLds[cur][1][nB][ks + 16 * g];
      b.q[1] = *(const uint4*)&bLds[cur][1][nB][ks + 16 * g + 8];
      WMMA16(acc31, a.v, b.v);
    }
    __syncthreads();  // everyone done reading buf[cur] before it is reused
  }

  const int nbase0 = n0 + nc0 * 16 + lm;
  const int nbase1 = nbase0 + 16;
#pragma unroll
  for (int r = 0; r < 8; ++r) {
    int ml = mrow * 16 + g * 8 + r;
    if (m0 + ml < cnt) {
      size_t rowoff = (size_t)(rowbase + m0 + ml) * HIDSZ;
      float g1 = acc10[r], g3 = acc30[r];
      Hout[rowoff + nbase0] = (_Float16)((g1 / (1.f + __expf(-g1))) * g3);
      g1 = acc11[r]; g3 = acc31[r];
      Hout[rowoff + nbase1] = (_Float16)((g1 / (1.f + __expf(-g1))) * g3);
    }
  }
}

// ------------------------------- GEMM2 -------------------------------------
// y = h @ W2; both A (contiguous slot rows of h) and B (pre-transposed W2)
// tiles staged by TDM. Shared expert (z==8) writes d_out directly.
__global__ __launch_bounds__(256)
void gemm2_kernel(const _Float16* __restrict__ h_rt,
                  const _Float16* __restrict__ h_sh,
                  const _Float16* __restrict__ w2t,
                  const _Float16* __restrict__ sw2t,
                  const int* __restrict__ counts,
                  const int* __restrict__ offs,
                  float* __restrict__ y_rt,
                  float* __restrict__ out) {
  const int e = blockIdx.z;
  const bool is_shared = (e == NEXP);
  const int cnt = is_shared ? T_TOK : counts[e];
  const int m0 = blockIdx.y * 64;
  if (m0 >= cnt) return;
  const int n0 = blockIdx.x * 64;
  const int rowbase = is_shared ? 0 : offs[e];
  const _Float16* Hsrc = is_shared ? h_sh : h_rt;
  const _Float16* W2 = is_shared ? sw2t : (w2t + (size_t)e * DIMSZ * HIDSZ);  // [D][H]
  float* Y = is_shared ? out : y_rt;

  __shared__ __align__(16) _Float16 aLds[2][64][64];
  __shared__ __align__(16) _Float16 bLds[2][64][64];

  const int tid = threadIdx.x;
  const int lane = tid & 31;
  const int wv = tid >> 5;
  const int mrow = wv >> 1;
  const int nc0 = (wv & 1) * 2;
  const int g = lane >> 4;
  const int lm = lane & 15;

  v8f acc0 = {}, acc1 = {};

  const int NCH = HIDSZ / 64;
  const _Float16* Abase = Hsrc + (size_t)(rowbase + m0) * HIDSZ;

  auto issue = [&](int kc, int buf) {
    int k0 = kc * 64;
    // rows past cnt-m0 read back zeros (remaining-extent OOB masking)
    tdm_load_2d(lds_addr_of(&aLds[buf][0][0]), Abase + k0,
                HIDSZ - k0, (unsigned)(cnt - m0), HIDSZ, 64, 64);
    tdm_load_2d(lds_addr_of(&bLds[buf][0][0]), W2 + (size_t)n0 * HIDSZ + k0,
                HIDSZ - k0, DIMSZ - n0, HIDSZ, 64, 64);
  };

  if (wv == 0) issue(0, 0);

  for (int i = 0; i < NCH; ++i) {
    const int cur = i & 1;
    if (i + 1 < NCH) {
      if (wv == 0) issue(i + 1, cur ^ 1);
      __builtin_amdgcn_s_wait_tensorcnt((short)2);
    } else {
      __builtin_amdgcn_s_wait_tensorcnt((short)0);
    }
    __syncthreads();

#pragma unroll
    for (int ks = 0; ks < 64; ks += 32) {
      FragA a, b;
      a.q[0] = *(const uint4*)&aLds[cur][mrow * 16 + lm][ks + 16 * g];
      a.q[1] = *(const uint4*)&aLds[cur][mrow * 16 + lm][ks + 16 * g + 8];
      const int nA = nc0 * 16 + lm;
      const int nB = nA + 16;
      b.q[0] = *(const uint4*)&bLds[cur][nA][ks + 16 * g];
      b.q[1] = *(const uint4*)&bLds[cur][nA][ks + 16 * g + 8];
      WMMA16(acc0, a.v, b.v);
      b.q[0] = *(const uint4*)&bLds[cur][nB][ks + 16 * g];
      b.q[1] = *(const uint4*)&bLds[cur][nB][ks + 16 * g + 8];
      WMMA16(acc1, a.v, b.v);
    }
    __syncthreads();
  }

  const int nbase0 = n0 + nc0 * 16 + lm;
  const int nbase1 = nbase0 + 16;
#pragma unroll
  for (int r = 0; r < 8; ++r) {
    int ml = mrow * 16 + g * 8 + r;
    if (m0 + ml < cnt) {
      size_t rowoff = (size_t)(rowbase + m0 + ml) * DIMSZ;
      Y[rowoff + nbase0] = acc0[r];
      Y[rowoff + nbase1] = acc1[r];
    }
  }
}

// ------------------------------ combine ------------------------------------
__global__ void combine_kernel(const float* __restrict__ y_rt,
                               const int* __restrict__ slot_of,
                               const float* __restrict__ gates,
                               float* __restrict__ out) {
  int i = blockIdx.x * blockDim.x + threadIdx.x;
  if (i >= T_TOK * DIMSZ) return;
  int t = i / DIMSZ, d = i - t * DIMSZ;
  int s0 = slot_of[2 * t], s1 = slot_of[2 * t + 1];
  out[i] += gates[2 * t]     * y_rt[(size_t)s0 * DIMSZ + d] +
            gates[2 * t + 1] * y_rt[(size_t)s1 * DIMSZ + d];
}

// ------------------------------ launcher -----------------------------------
extern "C" void kernel_launch(void* const* d_in, const int* in_sizes, int n_in,
                              void* d_out, int out_size, void* d_ws, size_t ws_size,
                              hipStream_t stream) {
  const float* x    = (const float*)d_in[0];
  const float* gw   = (const float*)d_in[1];
  const float* bias = (const float*)d_in[2];
  const float* w1   = (const float*)d_in[3];
  const float* w3   = (const float*)d_in[4];
  const float* w2   = (const float*)d_in[5];
  const float* sw1  = (const float*)d_in[6];
  const float* sw3  = (const float*)d_in[7];
  const float* sw2  = (const float*)d_in[8];
  float* out = (float*)d_out;

  char* base = (char*)d_ws;
  size_t off = 0;
  auto alloc = [&](size_t bytes) -> char* {
    char* p = base + off;
    off = (off + bytes + 255) & ~(size_t)255;
    return p;
  };

  _Float16* xh   = (_Float16*)alloc((size_t)T_TOK * DIMSZ * 2);
  _Float16* w1t  = (_Float16*)alloc((size_t)NEXP * DIMSZ * HIDSZ * 2);  // [E][H][D]
  _Float16* w3t  = (_Float16*)alloc((size_t)NEXP * DIMSZ * HIDSZ * 2);  // [E][H][D]
  _Float16* w2t  = (_Float16*)alloc((size_t)NEXP * HIDSZ * DIMSZ * 2);  // [E][D][H]
  _Float16* sw1t = (_Float16*)alloc((size_t)DIMSZ * HIDSZ * 2);          // [H][D]
  _Float16* sw3t = (_Float16*)alloc((size_t)DIMSZ * HIDSZ * 2);          // [H][D]
  _Float16* sw2t = (_Float16*)alloc((size_t)HIDSZ * DIMSZ * 2);          // [D][H]
  _Float16* h_rt = (_Float16*)alloc((size_t)T_TOK * TOPK * HIDSZ * 2);
  _Float16* h_sh = (_Float16*)alloc((size_t)T_TOK * HIDSZ * 2);
  float*    y_rt = (float*)   alloc((size_t)T_TOK * TOPK * DIMSZ * 4);
  int*   idx        = (int*)  alloc((size_t)T_TOK * TOPK * 4);
  float* gates      = (float*)alloc((size_t)T_TOK * TOPK * 4);
  int*   slot_of    = (int*)  alloc((size_t)T_TOK * TOPK * 4);
  int*   bucket_tok = (int*)  alloc((size_t)T_TOK * TOPK * 4);
  int*   counts     = (int*)  alloc(64);
  int*   offs       = (int*)  alloc(64);
  int*   cursor     = (int*)  alloc(64);
  (void)ws_size; (void)in_sizes; (void)n_in; (void)out_size;

  const int CT = 256;
  cvt_f32_f16<<<dim3((T_TOK * DIMSZ) / CT), dim3(CT), 0, stream>>>(
      x, xh, T_TOK * DIMSZ);

  dim3 tb(32, 8);
  transpose_cvt<<<dim3(HIDSZ / 32, DIMSZ / 32, NEXP), tb, 0, stream>>>(w1, w1t, DIMSZ, HIDSZ);
  transpose_cvt<<<dim3(HIDSZ / 32, DIMSZ / 32, NEXP), tb, 0, stream>>>(w3, w3t, DIMSZ, HIDSZ);
  transpose_cvt<<<dim3(DIMSZ / 32, HIDSZ / 32, NEXP), tb, 0, stream>>>(w2, w2t, HIDSZ, DIMSZ);
  transpose_cvt<<<dim3(HIDSZ / 32, DIMSZ / 32, 1),    tb, 0, stream>>>(sw1, sw1t, DIMSZ, HIDSZ);
  transpose_cvt<<<dim3(HIDSZ / 32, DIMSZ / 32, 1),    tb, 0, stream>>>(sw3, sw3t, DIMSZ, HIDSZ);
  transpose_cvt<<<dim3(DIMSZ / 32, HIDSZ / 32, 1),    tb, 0, stream>>>(sw2, sw2t, HIDSZ, DIMSZ);

  init_counts<<<1, 32, 0, stream>>>(counts);
  router_kernel<<<dim3(T_TOK / CT), dim3(CT), 0, stream>>>(x, gw, bias, idx, gates, counts);
  scan_kernel<<<1, 32, 0, stream>>>(counts, offs, cursor);
  scatter_kernel<<<dim3(T_TOK / CT), dim3(CT), 0, stream>>>(idx, cursor, bucket_tok, slot_of);

  gemm1_kernel<<<dim3(HIDSZ / 64, T_TOK / 64, NEXP + 1), dim3(256), 0, stream>>>(
      xh, w1t, w3t, sw1t, sw3t, counts, offs, bucket_tok, h_rt, h_sh);
  gemm2_kernel<<<dim3(DIMSZ / 64, T_TOK / 64, NEXP + 1), dim3(256), 0, stream>>>(
      h_rt, h_sh, w2t, sw2t, counts, offs, y_rt, out);

  combine_kernel<<<dim3((T_TOK * DIMSZ) / CT), dim3(CT), 0, stream>>>(
      y_rt, slot_of, gates, out);
}
